// Attention_9225589751981
// MI455X (gfx1250) — matrix operand order repaired and checked by reading, not compiled, not run on hardware
//
#include <hip/hip_runtime.h>

typedef _Float16 f16;
typedef __attribute__((ext_vector_type(16))) _Float16 v16h;
typedef __attribute__((ext_vector_type(8)))  _Float16 v8h;
typedef __attribute__((ext_vector_type(8)))  float    v8f;
typedef __attribute__((ext_vector_type(4)))  int      s4i;
typedef __attribute__((ext_vector_type(8)))  int      s8i;

#define DIM     768
#define NHEADS  12
#define HDIM    64
#define SEQ     1024
#define BATCH   8
#define ROWS    (BATCH * SEQ)   /* 8192 */
#define QKV_N   (3 * DIM)       /* 2304 */

__device__ __forceinline__ v16h cat16(v8h lo, v8h hi) {
  return __builtin_shufflevector(lo, hi, 0, 1, 2, 3, 4, 5, 6, 7,
                                 8, 9, 10, 11, 12, 13, 14, 15);
}

// ------------------------------------------------------------------ TDM
// 2-D f16 tile load (rows x cols, row stride = stride_elems) from global
// into LDS at lds_off, with LDS row padding described by pad codes.
// pad_interval_code: 0=2,1=4,2=8,3=16,4=32,... DWORDs between pads
// pad_amount_code:   0=1,1=2,...  DWORDs of pad inserted
__device__ __forceinline__ void tdm_load_2d_f16(
    unsigned lds_off, const f16* gptr, int stride_elems,
    int rows, int cols, int pad_interval_code, int pad_amount_code) {
  unsigned long long ga = (unsigned long long)(size_t)gptr;
  s4i g0;
  g0[0] = 1;                                        // count=1, user mode
  g0[1] = (int)lds_off;                             // lds_addr
  g0[2] = (int)(ga & 0xffffffffull);                // global_addr[31:0]
  g0[3] = (int)((ga >> 32) & 0x1ffffffull) | (2 << 30);  // addr[56:32], type=2
  s8i g1;
  g1[0] = (1 << 16) /*data_size=2B*/ | (1 << 20) /*pad_enable*/ |
          (pad_interval_code << 22) | (pad_amount_code << 25);
  g1[1] = (cols & 0xffff) << 16;                    // tensor_dim0 lo16
  g1[2] = ((cols >> 16) & 0xffff) | ((rows & 0xffff) << 16);  // dim0 hi / dim1 lo
  g1[3] = ((rows >> 16) & 0xffff) | ((cols & 0xffff) << 16);  // dim1 hi / tile_dim0
  g1[4] = rows & 0xffff;                            // tile_dim1
  g1[5] = stride_elems;                             // tensor_dim0_stride lo32
  g1[6] = 0;
  g1[7] = 0;
  asm volatile("tensor_load_to_lds %0, %1" :: "s"(g0), "s"(g1) : "memory");
}

// ---------------------------------------------------------------- convert
__global__ void cvt_f32_f16(const float* __restrict__ src,
                            f16* __restrict__ dst, int n) {
  int i = blockIdx.x * blockDim.x + threadIdx.x;
  int stride = gridDim.x * blockDim.x;
  for (; i < n; i += stride) dst[i] = (f16)src[i];
}

// ---------------------------------------------------------------- GEMM
// C[M,N] = A[M,K] (f16, row-major) * B[K,N] (f16, row-major) + bias (f32)
// Double-buffered LDS; A tiles via async TDM, B tiles via coalesced loads +
// transpose scatter. One barrier per K-step; TDM latency hidden by compute.
// OUT_MODE 0: f32 row-major store.
// OUT_MODE 1: f16 store scattered into qkv layout [3][B][H][SEQ][HDIM].
template <int OUT_MODE>
__global__ void __launch_bounds__(256)
gemm_f16(const f16* __restrict__ A, const f16* __restrict__ B,
         const float* __restrict__ bias, void* __restrict__ outp,
         int M, int N, int K) {
  constexpr int BM = 128, BN = 128, BK = 32;
  __shared__ __align__(16) f16 sA[2][BM][BK + 8];   // row: 16 dw + 4 dw pad
  __shared__ __align__(16) f16 sB[2][BN][BK + 8];   // transposed [n][k]

  const int tid   = threadIdx.x;
  const int lane  = tid & 31;
  const int wave  = tid >> 5;     // 0..7
  const int wm    = wave >> 1;    // 0..3  (32 rows each)
  const int wn    = wave & 1;     // 0..1  (64 cols each)
  const int lrow  = lane & 15;
  const int lhalf = lane >> 4;
  const int bm0 = blockIdx.y * BM;
  const int bn0 = blockIdx.x * BN;

  v8f acc[2][4] = {};

  auto issueA = [&](int k0, int buf) {
    if (wave == 0)
      tdm_load_2d_f16((unsigned)(size_t)&sA[buf][0][0],
                      A + (size_t)bm0 * K + k0, K, BM, BK,
                      /*interval 16dw*/ 3, /*amount 4dw*/ 3);
  };
  auto stageB = [&](int k0, int buf) {
#pragma unroll
    for (int it = 0; it < 2; ++it) {
      int idx = tid + it * 256;            // 0..511
      int k   = idx >> 4;                  // 0..31
      int nc  = (idx & 15) * 8;            // 0..120
      v8h bv = *(const v8h*)(B + (size_t)(k0 + k) * N + bn0 + nc);
#pragma unroll
      for (int j = 0; j < 8; ++j) sB[buf][nc + j][k] = bv[j];
    }
  };

  // prologue: fill buffer 0
  issueA(0, 0);
  stageB(0, 0);
  if (wave == 0) __builtin_amdgcn_s_wait_tensorcnt(0);
  __syncthreads();

  const int nk = K / BK;
  for (int ki = 0; ki < nk; ++ki) {
    const int buf = ki & 1;
    // prefetch next tile into the other buffer while we compute this one
    if (ki + 1 < nk) {
      issueA((ki + 1) * BK, buf ^ 1);
      stageB((ki + 1) * BK, buf ^ 1);
    }

    v16h afr[2], bfr[4];
#pragma unroll
    for (int mi = 0; mi < 2; ++mi) {
      const f16* p = &sA[buf][wm * 32 + mi * 16 + lrow][lhalf * 8];
      afr[mi] = cat16(*(const v8h*)p, *(const v8h*)(p + 16));
    }
#pragma unroll
    for (int ni = 0; ni < 4; ++ni) {
      const f16* p = &sB[buf][wn * 64 + ni * 16 + lrow][lhalf * 8];
      bfr[ni] = cat16(*(const v8h*)p, *(const v8h*)(p + 16));
    }
#pragma unroll
    for (int mi = 0; mi < 2; ++mi)
#pragma unroll
      for (int ni = 0; ni < 4; ++ni)
        acc[mi][ni] = __builtin_amdgcn_wmma_f32_16x16x32_f16(
            false, afr[mi], false, bfr[ni], (short)0, acc[mi][ni],
            false, false);

    if (wave == 0) __builtin_amdgcn_s_wait_tensorcnt(0);
    __syncthreads();
  }

  // epilogue
#pragma unroll
  for (int mi = 0; mi < 2; ++mi) {
#pragma unroll
    for (int ni = 0; ni < 4; ++ni) {
      int col = bn0 + wn * 64 + ni * 16 + lrow;
      float bv = bias[col];
#pragma unroll
      for (int i = 0; i < 8; ++i) {
        int row = bm0 + wm * 32 + mi * 16 + lhalf * 8 + i;
        float v = acc[mi][ni][i] + bv;
        if (OUT_MODE == 0) {
          ((float*)outp)[(size_t)row * N + col] = v;
        } else {
          int which = col / DIM;          // 0=q,1=k,2=v
          int rem   = col - which * DIM;
          int h = rem >> 6;
          int d = rem & 63;
          int b  = row >> 10;
          int nq = row & 1023;
          size_t off =
              ((((size_t)which * BATCH + b) * NHEADS + h) * SEQ + nq) * HDIM + d;
          ((f16*)outp)[off] = (f16)v;
        }
      }
    }
  }
}

// ---------------------------------------------------------------- flash attention
// qkv: [3][B][H][SEQ][HDIM] f16. out: [B][SEQ][DIM] f16 (proj input layout).
// One block per (b, h, 128-query slab); wave owns 16 queries; 64-key chunks.
// Double-buffered K (TDM) / V (transpose-scatter) staging, 1 barrier/chunk.
__global__ void __launch_bounds__(256)
attn_kernel(const f16* __restrict__ qkv, f16* __restrict__ out) {
  constexpr float SCALE_ = 0.125f;  // HDIM^-0.5
  constexpr int KB = 64;            // keys per chunk
  __shared__ __align__(16) f16 sK[2][KB][HDIM + 8];   // 32 dw row + 4 dw pad
  __shared__ __align__(16) f16 sV[2][HDIM][KB + 8];   // transposed [d][k]
  __shared__ __align__(16) f16 sP[8][16][KB + 8];     // per-wave P scratch

  const int tid   = threadIdx.x;
  const int lane  = tid & 31;
  const int wave  = tid >> 5;
  const int lrow  = lane & 15;
  const int lhalf = lane >> 4;

  const int qblk = blockIdx.x & 7;
  const int bh   = blockIdx.x >> 3;   // 0..95
  const size_t head_off = (size_t)bh * SEQ * HDIM;
  const size_t plane    = (size_t)BATCH * NHEADS * SEQ * HDIM;
  const f16* Q  = qkv + head_off;
  const f16* Kp = qkv + plane + head_off;
  const f16* Vp = qkv + 2 * plane + head_off;

  const int q0 = qblk * 128 + wave * 16;

  auto issueK = [&](int kc, int buf) {
    if (wave == 0)
      tdm_load_2d_f16((unsigned)(size_t)&sK[buf][0][0],
                      Kp + (size_t)kc * HDIM, HDIM, KB, HDIM,
                      /*interval 32dw*/ 4, /*amount 4dw*/ 3);
  };
  auto stageV = [&](int kc, int buf) {
#pragma unroll
    for (int it = 0; it < 2; ++it) {
      int idx = tid + it * 256;        // 0..511
      int k   = idx >> 3;              // 0..63
      int dc  = (idx & 7) * 8;         // 0..56
      v8h vv = *(const v8h*)(Vp + (size_t)(kc + k) * HDIM + dc);
#pragma unroll
      for (int j = 0; j < 8; ++j) sV[buf][dc + j][k] = vv[j];
    }
  };

  // Q fragments for dim-chunks 0..31 / 32..63
  v16h aQ[2];
#pragma unroll
  for (int kk = 0; kk < 2; ++kk) {
    const f16* p = Q + (size_t)(q0 + lrow) * HDIM + kk * 32 + lhalf * 8;
    aQ[kk] = cat16(*(const v8h*)p, *(const v8h*)(p + 16));
  }

  v8f o[4] = {};
  float mrun[8], lrun[8];
#pragma unroll
  for (int i = 0; i < 8; ++i) { mrun[i] = -3.0e38f; lrun[i] = 0.f; }

  // prologue: fill buffer 0
  issueK(0, 0);
  stageV(0, 0);
  if (wave == 0) __builtin_amdgcn_s_wait_tensorcnt(0);
  __syncthreads();

  constexpr int NCHUNK = SEQ / KB;   // 16
  for (int ci = 0; ci < NCHUNK; ++ci) {
    const int buf = ci & 1;
    // prefetch next chunk into the other buffer
    if (ci + 1 < NCHUNK) {
      issueK((ci + 1) * KB, buf ^ 1);
      stageV((ci + 1) * KB, buf ^ 1);
    }

    // scores: S = Q (16x64) @ K^T (64x64) -> four 16x16 C tiles
    v8f s[4];
#pragma unroll
    for (int ni = 0; ni < 4; ++ni) {
      v8f z = {};
#pragma unroll
      for (int kk = 0; kk < 2; ++kk) {
        const f16* p = &sK[buf][ni * 16 + lrow][kk * 32 + lhalf * 8];
        v16h bfr = cat16(*(const v8h*)p, *(const v8h*)(p + 16));
        z = __builtin_amdgcn_wmma_f32_16x16x32_f16(
            false, aQ[kk], false, bfr, (short)0, z, false, false);
      }
#pragma unroll
      for (int i = 0; i < 8; ++i) z[i] *= SCALE_;
      s[ni] = z;
    }

    // online softmax (row i lives in 16-lane half, VGPR i)
    float corr[8];
#pragma unroll
    for (int i = 0; i < 8; ++i) {
      float mx = fmaxf(fmaxf(s[0][i], s[1][i]), fmaxf(s[2][i], s[3][i]));
#pragma unroll
      for (int off = 1; off < 16; off <<= 1)
        mx = fmaxf(mx, __shfl_xor(mx, off, 32));
      float mnew = fmaxf(mrun[i], mx);
      corr[i] = __expf(mrun[i] - mnew);
      mrun[i] = mnew;
      float rs = 0.f;
#pragma unroll
      for (int ni = 0; ni < 4; ++ni) {
        float p = __expf(s[ni][i] - mnew);
        s[ni][i] = p;
        rs += p;
      }
#pragma unroll
      for (int off = 1; off < 16; off <<= 1)
        rs += __shfl_xor(rs, off, 32);
      lrun[i] = lrun[i] * corr[i] + rs;
    }
#pragma unroll
    for (int j = 0; j < 4; ++j)
#pragma unroll
      for (int i = 0; i < 8; ++i) o[j][i] *= corr[i];

    // C-layout -> A-layout for P via per-wave LDS round trip
#pragma unroll
    for (int ni = 0; ni < 4; ++ni)
#pragma unroll
      for (int i = 0; i < 8; ++i)
        sP[wave][lhalf * 8 + i][ni * 16 + lrow] = (f16)s[ni][i];

    v16h aP[2];
#pragma unroll
    for (int kb = 0; kb < 2; ++kb) {
      const f16* pp = &sP[wave][lrow][kb * 32 + lhalf * 8];
      aP[kb] = cat16(*(const v8h*)pp, *(const v8h*)(pp + 16));
    }

    // O += P (16x64) @ V (64x64)
#pragma unroll
    for (int j = 0; j < 4; ++j) {
#pragma unroll
      for (int kb = 0; kb < 2; ++kb) {
        const f16* p = &sV[buf][j * 16 + lrow][kb * 32 + lhalf * 8];
        v16h bfr = cat16(*(const v8h*)p, *(const v8h*)(p + 16));
        o[j] = __builtin_amdgcn_wmma_f32_16x16x32_f16(
            false, aP[kb], false, bfr, (short)0, o[j], false, false);
      }
    }

    if (wave == 0) __builtin_amdgcn_s_wait_tensorcnt(0);
    __syncthreads();
  }

  // finalize: divide by l, store f16 at [b][q][h*64+d]
  const int b = bh / NHEADS;
  const int h = bh % NHEADS;
#pragma unroll
  for (int j = 0; j < 4; ++j) {
    int d = j * 16 + lrow;
#pragma unroll
    for (int i = 0; i < 8; ++i) {
      int q = q0 + lhalf * 8 + i;
      float v = o[j][i] / lrun[i];
      out[((size_t)(b * SEQ + q)) * DIM + h * HDIM + d] = (f16)v;
    }
  }
}

// ---------------------------------------------------------------- launch
extern "C" void kernel_launch(void* const* d_in, const int* in_sizes, int n_in,
                              void* d_out, int out_size, void* d_ws, size_t ws_size,
                              hipStream_t stream) {
  const float* x      = (const float*)d_in[0];
  const float* w_qkv  = (const float*)d_in[1];
  const float* b_qkv  = (const float*)d_in[2];
  const float* w_proj = (const float*)d_in[3];
  const float* b_proj = (const float*)d_in[4];

  char* ws = (char*)d_ws;
  size_t off = 0;
  auto alloc = [&](size_t bytes) -> void* {
    void* p = ws + off;
    off += (bytes + 255) & ~(size_t)255;
    return p;
  };
  f16* xb    = (f16*)alloc((size_t)ROWS * DIM * sizeof(f16));
  f16* wqb   = (f16*)alloc((size_t)DIM * QKV_N * sizeof(f16));
  f16* wpb   = (f16*)alloc((size_t)DIM * DIM * sizeof(f16));
  f16* qkv   = (f16*)alloc((size_t)3 * ROWS * DIM * sizeof(f16));
  f16* attno = (f16*)alloc((size_t)ROWS * DIM * sizeof(f16));

  cvt_f32_f16<<<512, 256, 0, stream>>>(x, xb, ROWS * DIM);
  cvt_f32_f16<<<512, 256, 0, stream>>>(w_qkv, wqb, DIM * QKV_N);
  cvt_f32_f16<<<256, 256, 0, stream>>>(w_proj, wpb, DIM * DIM);

  dim3 g1(QKV_N / 128, ROWS / 128);
  gemm_f16<1><<<g1, 256, 0, stream>>>(xb, wqb, b_qkv, (void*)qkv,
                                      ROWS, QKV_N, DIM);

  attn_kernel<<<BATCH * NHEADS * 8, 256, 0, stream>>>(qkv, attno);

  dim3 g2(DIM / 128, ROWS / 128);
  gemm_f16<0><<<g2, 256, 0, stream>>>(attno, wpb, b_proj, d_out,
                                      ROWS, DIM, DIM);
}
